// Encoder_35931696398993
// MI455X (gfx1250) — matrix operand order repaired and checked by reading, not compiled, not run on hardware
//
#include <hip/hip_runtime.h>

typedef unsigned short u16;
typedef unsigned int   u32;

typedef __bf16 v16bf __attribute__((ext_vector_type(16)));
typedef float  v8f   __attribute__((ext_vector_type(8)));

union frag16 { uint4 q[2]; v16bf v; };

__device__ __forceinline__ u16 f2bf(float f) {
    union { float f; u32 u; } v; v.f = f;
    u32 r = v.u + 0x7FFFu + ((v.u >> 16) & 1u);   // round-to-nearest-even
    return (u16)(r >> 16);
}
__device__ __forceinline__ float bf2f(u16 h) {
    union { u32 u; float f; } v; v.u = ((u32)h) << 16;
    return v.f;
}

// ---- CDNA5 async global->LDS copy (16B per lane), tracked by ASYNCcnt ----
__device__ __forceinline__ void async_copy16(const void* gptr, void* lptr) {
    u32 lds = (u32)(unsigned long long)lptr;   // LDS aperture: offset in low 32 bits
    asm volatile("global_load_async_to_lds_b128 %0, %1, off"
                 :: "v"(lds), "v"(gptr) : "memory");
}
__device__ __forceinline__ void wait_async() {
    asm volatile("s_wait_asynccnt 0" ::: "memory");
}

// ---------------------------------------------------------------------------
// x (f32) -> bf16 elementwise
// ---------------------------------------------------------------------------
__global__ __launch_bounds__(256) void cvt_kernel(
    const float* __restrict__ s, u16* __restrict__ d, int n)
{
    int i = blockIdx.x * 256 + threadIdx.x;
    if (i < n) d[i] = f2bf(s[i]);
}

// ---------------------------------------------------------------------------
// Wt[n][k] (bf16) = W[k][n] (f32): pre-transpose weights into the N-major
// layout the WMMA B-fragment wants, so GEMM tile fills are straight copies.
// Grid: (N/32, K/32), block 256 (32x8).
// ---------------------------------------------------------------------------
__global__ __launch_bounds__(256) void tr_kernel(
    const float* __restrict__ W, u16* __restrict__ Wt, int K, int N)
{
    __shared__ float tile[32][33];
    const int n0 = blockIdx.x * 32, k0 = blockIdx.y * 32;
    const int tx = threadIdx.x & 31, ty = threadIdx.x >> 5;
    for (int i = ty; i < 32; i += 8)
        tile[i][tx] = W[(size_t)(k0 + i) * N + n0 + tx];
    __syncthreads();
    for (int i = ty; i < 32; i += 8)
        Wt[(size_t)(n0 + i) * K + k0 + tx] = f2bf(tile[tx][i]);
}

// ---------------------------------------------------------------------------
// GEMM: C[M,N] = A[M,K] @ Wt[N,K]^T + bias, A/Wt bf16, double-buffered LDS
// tiles filled with global_load_async_to_lds_b128. Outputs f32 and/or bf16.
// Block = 256 threads (8 waves); BM=BN=128, BK=32; wave owns 32x64 (2x4 tiles).
// ---------------------------------------------------------------------------
__device__ __forceinline__ void gemm_issue_tile(
    const u16* __restrict__ A, const u16* __restrict__ W,
    u16* lA, u16* lB, int m0, int n0, int K, int k0, int t)
{
    for (int p = 0; p < 2; ++p) {
        int chunk = p * 256 + t;              // 0..511; 128 rows x 4 chunks
        int r = chunk >> 2, off = (chunk & 3) * 8;
        async_copy16(A + (size_t)(m0 + r) * K + k0 + off, lA + r * 32 + off);
        async_copy16(W + (size_t)(n0 + r) * K + k0 + off, lB + r * 32 + off);
    }
}

__global__ __launch_bounds__(256) void gemm_bf16_kernel(
    const u16* __restrict__ A, const u16* __restrict__ W,
    const float* __restrict__ bias, float* __restrict__ Cf,
    u16* __restrict__ Cb, int M, int N, int K, int relu)
{
    __shared__ u16 lA[2][128 * 32];
    __shared__ u16 lB[2][128 * 32];

    const int t     = threadIdx.x;
    const int lane  = t & 31;
    const int wave  = t >> 5;
    const int l16   = lane & 15;
    const int lhalf = (lane >> 4) & 1;
    const int m0 = blockIdx.y * 128;
    const int n0 = blockIdx.x * 128;
    const int wm = wave >> 1;   // 0..3
    const int wn = wave & 1;    // 0..1
    (void)M;

    v8f acc[2][4];
    for (int i = 0; i < 2; ++i)
        for (int j = 0; j < 4; ++j)
            acc[i][j] = (v8f){};

    gemm_issue_tile(A, W, lA[0], lB[0], m0, n0, K, 0, t);

    int buf = 0;
    for (int k0 = 0; k0 < K; k0 += 32, buf ^= 1) {
        wait_async();
        __syncthreads();                       // tiles landed; prev reads done
        if (k0 + 32 < K)
            gemm_issue_tile(A, W, lA[buf ^ 1], lB[buf ^ 1], m0, n0, K, k0 + 32, t);

        frag16 af[2], bf[4];
        for (int i = 0; i < 2; ++i) {
            const u16* ap = &lA[buf][(wm * 32 + i * 16 + l16) * 32 + (lhalf ? 8 : 0)];
            af[i].q[0] = *reinterpret_cast<const uint4*>(ap);
            af[i].q[1] = *reinterpret_cast<const uint4*>(ap + 16);
        }
        for (int j = 0; j < 4; ++j) {
            const u16* bp = &lB[buf][(wn * 64 + j * 16 + l16) * 32 + (lhalf ? 8 : 0)];
            bf[j].q[0] = *reinterpret_cast<const uint4*>(bp);
            bf[j].q[1] = *reinterpret_cast<const uint4*>(bp + 16);
        }
        for (int i = 0; i < 2; ++i)
            for (int j = 0; j < 4; ++j)
                acc[i][j] = __builtin_amdgcn_wmma_f32_16x16x32_bf16(
                    false, af[i].v, false, bf[j].v, (short)0, acc[i][j],
                    false, false);
    }

    for (int i = 0; i < 2; ++i) {
        int rowb = m0 + wm * 32 + i * 16 + lhalf * 8;
        for (int j = 0; j < 4; ++j) {
            int col = n0 + wn * 64 + j * 16 + l16;
            float bv = bias ? bias[col] : 0.0f;
            for (int e = 0; e < 8; ++e) {
                float v = acc[i][j][e] + bv;
                if (relu) v = fmaxf(v, 0.0f);
                if (Cf) Cf[(size_t)(rowb + e) * N + col] = v;
                if (Cb) Cb[(size_t)(rowb + e) * N + col] = f2bf(v);
            }
        }
    }
}

// ---------------------------------------------------------------------------
// Fused dual-map flash attention (bf16 Q/K/V in, bf16 ctx out).
// (Q+r)(K+r)^T = QK^T + Q.r + K.r + r.r  -> single QK^T WMMA per tile plus
// f32 rank-1 corrections folded into the softmax pass.
// Grid: (L/64, B*H). Block 256 (8 waves). 32-key chunks, online softmax,
// post-softmax masks folded into unnormalized bf16 P, PV via WMMA with
// per-row rescale of persistent accumulators.
// ---------------------------------------------------------------------------
__global__ __launch_bounds__(256) void attn_kernel(
    const u16* __restrict__ Qg, const u16* __restrict__ Kg,
    const u16* __restrict__ Vg, const float* __restrict__ rel,
    const float* __restrict__ adjm, const float* __restrict__ comm,
    u16* __restrict__ ctxb)
{
    __shared__ u16   Qs[64 * 64];      // [q][d]
    __shared__ u16   Ks[32 * 64];      // [key][d] (N-major for S)
    __shared__ u16   Vt[64 * 32];      // [d][key] (N-major for PV)
    __shared__ float Sraw[64 * 36];    // QK^T tile, stride 36
    __shared__ u16   Pa[64 * 32], Pc[64 * 32];
    __shared__ float relS[128];
    __shared__ float qcorr[128];       // Q.r_adj[64], Q.r_com[64]
    __shared__ float kcorr[64];        // K.r_adj[32], K.r_com[32]
    __shared__ float rrs[2];           // r.r for adj/com
    __shared__ float stats[384];       // m_a z_a sc_a m_c z_c sc_c

    const int t     = threadIdx.x;
    const int lane  = t & 31;
    const int wave  = t >> 5;
    const int l16   = lane & 15;
    const int lhalf = (lane >> 4) & 1;
    const int bh = blockIdx.y;
    const int q0 = blockIdx.x * 64;
    const int wm = wave >> 1;
    const int wn = wave & 1;

    if (t < 128) relS[t] = rel[t];
    if (t < 64) {
        stats[t]       = -1e30f; stats[64 + t]  = 0.0f;
        stats[192 + t] = -1e30f; stats[256 + t] = 0.0f;
    }
    // async-load Q tile (64x64 bf16 = 512 x 16B chunks)
    for (int p = 0; p < 2; ++p) {
        int chunk = p * 256 + t;
        int r = chunk >> 3, off = (chunk & 7) * 8;
        async_copy16(Qg + (size_t)bh * 65536 + (size_t)(q0 + r) * 64 + off,
                     Qs + r * 64 + off);
    }
    wait_async();
    __syncthreads();

    // Q.r corrections + r.r
    {
        int row = t >> 2, qtr = (t & 3) * 16;
        float qa = 0.0f, qc = 0.0f;
        for (int i = 0; i < 16; ++i) {
            float qv = bf2f(Qs[row * 64 + qtr + i]);
            qa += qv * relS[qtr + i];
            qc += qv * relS[64 + qtr + i];
        }
        for (int o = 1; o < 4; o <<= 1) {
            qa += __shfl_xor(qa, o, 4);
            qc += __shfl_xor(qc, o, 4);
        }
        if ((t & 3) == 0) { qcorr[row] = qa; qcorr[64 + row] = qc; }
        if (t < 2) {
            float s = 0.0f;
            for (int i = 0; i < 64; ++i) s += relS[t * 64 + i] * relS[t * 64 + i];
            rrs[t] = s;
        }
    }

    v8f accA[2], accC[2];
    accA[0] = (v8f){}; accA[1] = (v8f){};
    accC[0] = (v8f){}; accC[1] = (v8f){};
    __syncthreads();

    for (int k0 = 0; k0 < 1024; k0 += 32) {
        // K chunk async (32x64 bf16 = 256 x 16B chunks); V transposed (u16 copy)
        {
            int r = t >> 3, off = (t & 7) * 8;
            async_copy16(Kg + (size_t)bh * 65536 + (size_t)(k0 + r) * 64 + off,
                         Ks + r * 64 + off);
        }
        for (int li = t; li < 2048; li += 256) {
            int r = li >> 6, d = li & 63;
            Vt[d * 32 + r] = Vg[(size_t)bh * 65536 + (size_t)(k0 + r) * 64 + d];
        }
        wait_async();
        __syncthreads();

        // S = QK^T (one 16x16 tile per wave), plus K.r corrections (waves 0-3)
        {
            const int mrow = wm * 16, ncol = wn * 16;
            v8f s = (v8f){};
            for (int kk = 0; kk < 64; kk += 32) {
                frag16 a, b;
                const u16* ap = &Qs[(mrow + l16) * 64 + kk + (lhalf ? 8 : 0)];
                const u16* bp = &Ks[(ncol + l16) * 64 + kk + (lhalf ? 8 : 0)];
                a.q[0] = *reinterpret_cast<const uint4*>(ap);
                a.q[1] = *reinterpret_cast<const uint4*>(ap + 16);
                b.q[0] = *reinterpret_cast<const uint4*>(bp);
                b.q[1] = *reinterpret_cast<const uint4*>(bp + 16);
                s = __builtin_amdgcn_wmma_f32_16x16x32_bf16(
                        false, a.v, false, b.v, (short)0, s, false, false);
            }
            int colg = ncol + l16, rowb = mrow + lhalf * 8;
            for (int e = 0; e < 8; ++e)
                Sraw[(rowb + e) * 36 + colg] = s[e];
        }
        if (t < 128) {
            int row = t >> 2, qtr = (t & 3) * 16;
            float ka = 0.0f, kc = 0.0f;
            for (int i = 0; i < 16; ++i) {
                float kv = bf2f(Ks[row * 64 + qtr + i]);
                ka += kv * relS[qtr + i];
                kc += kv * relS[64 + qtr + i];
            }
            for (int o = 1; o < 4; o <<= 1) {
                ka += __shfl_xor(ka, o, 4);
                kc += __shfl_xor(kc, o, 4);
            }
            if ((t & 3) == 0) { kcorr[row] = ka; kcorr[32 + row] = kc; }
        }
        __syncthreads();

        // online softmax for both maps; masks folded into unnormalized P
        {
            int row = t >> 2, seg = (t & 3) * 8;
            float qa = qcorr[row], qc = qcorr[64 + row];
            float sA[8], sC[8];
            for (int i = 0; i < 8; ++i) {
                float base = Sraw[row * 36 + seg + i];
                sA[i] = (base + qa + kcorr[seg + i] + rrs[0]) * 0.125f;
                sC[i] = (base + qc + kcorr[32 + seg + i] + rrs[1]) * 0.125f;
            }
            const float* mpA = adjm + (size_t)bh * 1048576 +
                               (size_t)(q0 + row) * 1024 + k0 + seg;
            const float* mpC = comm + (size_t)bh * 1048576 +
                               (size_t)(q0 + row) * 1024 + k0 + seg;
            if (k0 < 992) {                       // stream next mask chunk
                __builtin_prefetch(mpA + 32, 0, 1);
                __builtin_prefetch(mpC + 32, 0, 1);
            }
            // adj
            {
                float mo = stats[row];
                float bmax = -1e30f;
                for (int i = 0; i < 8; ++i) bmax = fmaxf(bmax, sA[i]);
                for (int o = 1; o < 4; o <<= 1)
                    bmax = fmaxf(bmax, __shfl_xor(bmax, o, 4));
                float mn = fmaxf(mo, bmax);
                float bs = 0.0f;
                for (int i = 0; i < 8; ++i) {
                    float p = __expf(sA[i] - mn);
                    bs += p;
                    Pa[row * 32 + seg + i] = f2bf(p * mpA[i]);
                }
                for (int o = 1; o < 4; o <<= 1) bs += __shfl_xor(bs, o, 4);
                if ((t & 3) == 0) {
                    float al = __expf(mo - mn);
                    stats[128 + row] = al;
                    stats[64 + row]  = stats[64 + row] * al + bs;
                    stats[row]       = mn;
                }
            }
            // com
            {
                float mo = stats[192 + row];
                float bmax = -1e30f;
                for (int i = 0; i < 8; ++i) bmax = fmaxf(bmax, sC[i]);
                for (int o = 1; o < 4; o <<= 1)
                    bmax = fmaxf(bmax, __shfl_xor(bmax, o, 4));
                float mn = fmaxf(mo, bmax);
                float bs = 0.0f;
                for (int i = 0; i < 8; ++i) {
                    float p = __expf(sC[i] - mn);
                    bs += p;
                    Pc[row * 32 + seg + i] = f2bf(p * mpC[i]);
                }
                for (int o = 1; o < 4; o <<= 1) bs += __shfl_xor(bs, o, 4);
                if ((t & 3) == 0) {
                    float al = __expf(mo - mn);
                    stats[320 + row] = al;
                    stats[256 + row] = stats[256 + row] * al + bs;
                    stats[192 + row] = mn;
                }
            }
        }
        __syncthreads();

        // rescale persistent accumulators + PV WMMA
        {
            int rowb = wm * 16 + lhalf * 8;
            float sa8[8], sc8[8];
            for (int e = 0; e < 8; ++e) {
                sa8[e] = stats[128 + rowb + e];
                sc8[e] = stats[320 + rowb + e];
            }
            for (int j = 0; j < 2; ++j)
                for (int e = 0; e < 8; ++e) {
                    accA[j][e] *= sa8[e];
                    accC[j][e] *= sc8[e];
                }
            frag16 a, ac;
            const u16* ap  = &Pa[(wm * 16 + l16) * 32 + (lhalf ? 8 : 0)];
            const u16* acp = &Pc[(wm * 16 + l16) * 32 + (lhalf ? 8 : 0)];
            a.q[0]  = *reinterpret_cast<const uint4*>(ap);
            a.q[1]  = *reinterpret_cast<const uint4*>(ap + 16);
            ac.q[0] = *reinterpret_cast<const uint4*>(acp);
            ac.q[1] = *reinterpret_cast<const uint4*>(acp + 16);
            for (int j = 0; j < 2; ++j) {
                frag16 b;
                const u16* bp = &Vt[((wn * 2 + j) * 16 + l16) * 32 + (lhalf ? 8 : 0)];
                b.q[0] = *reinterpret_cast<const uint4*>(bp);
                b.q[1] = *reinterpret_cast<const uint4*>(bp + 16);
                accA[j] = __builtin_amdgcn_wmma_f32_16x16x32_bf16(
                    false, a.v,  false, b.v, (short)0, accA[j], false, false);
                accC[j] = __builtin_amdgcn_wmma_f32_16x16x32_bf16(
                    false, ac.v, false, b.v, (short)0, accC[j], false, false);
            }
        }
        __syncthreads();
    }

    // final normalization, combine maps, write bf16 context
    {
        int rowb = wm * 16 + lhalf * 8;
        for (int j = 0; j < 2; ++j) {
            int col = (wn * 2 + j) * 16 + l16;
            for (int e = 0; e < 8; ++e) {
                int row = rowb + e;
                float v = accA[j][e] / stats[64 + row] +
                          accC[j][e] / stats[256 + row];
                ctxb[(size_t)bh * 65536 + (size_t)(q0 + row) * 64 + col] = f2bf(v);
            }
        }
    }
}

// ---------------------------------------------------------------------------
// LayerNorm(a + b) * g + beta ; optional f32 and bf16 outputs (D = 1024)
// ---------------------------------------------------------------------------
__global__ __launch_bounds__(256) void ln_residual_kernel(
    const float* __restrict__ a, const float* __restrict__ b,
    const float* __restrict__ g, const float* __restrict__ be,
    float* __restrict__ outf, u16* __restrict__ outb)
{
    __shared__ float red[256];
    __shared__ float mv[2];
    const int row = blockIdx.x, t = threadIdx.x;

    float v[4];
    float s = 0.0f;
    for (int i = 0; i < 4; ++i) {
        size_t idx = (size_t)row * 1024 + t + i * 256;
        v[i] = a[idx] + b[idx];
        s += v[i];
    }
    red[t] = s; __syncthreads();
    for (int o = 128; o > 0; o >>= 1) {
        if (t < o) red[t] += red[t + o];
        __syncthreads();
    }
    if (t == 0) mv[0] = red[0] * (1.0f / 1024.0f);
    __syncthreads();
    const float mean = mv[0];

    s = 0.0f;
    for (int i = 0; i < 4; ++i) { float d = v[i] - mean; s += d * d; }
    red[t] = s; __syncthreads();
    for (int o = 128; o > 0; o >>= 1) {
        if (t < o) red[t] += red[t + o];
        __syncthreads();
    }
    if (t == 0) mv[1] = red[0] * (1.0f / 1024.0f);
    __syncthreads();
    const float rstd = rsqrtf(mv[1] + 1e-5f);

    for (int i = 0; i < 4; ++i) {
        int c = t + i * 256;
        float r = (v[i] - mean) * rstd * g[c] + be[c];
        size_t idx = (size_t)row * 1024 + c;
        if (outf) outf[idx] = r;
        if (outb) outb[idx] = f2bf(r);
    }
}

// ---------------------------------------------------------------------------
extern "C" void kernel_launch(void* const* d_in, const int* in_sizes, int n_in,
                              void* d_out, int out_size, void* d_ws, size_t ws_size,
                              hipStream_t stream)
{
    (void)in_sizes; (void)n_in; (void)out_size; (void)ws_size;

    const float* x    = (const float*)d_in[0];
    const float* adjm = (const float*)d_in[1];
    const float* comm = (const float*)d_in[2];
    const float* rel  = (const float*)d_in[3];
    const float* Wq = (const float*)d_in[4];  const float* bq  = (const float*)d_in[5];
    const float* Wk = (const float*)d_in[6];  const float* bk  = (const float*)d_in[7];
    const float* Wv = (const float*)d_in[8];  const float* bv  = (const float*)d_in[9];
    const float* Wo = (const float*)d_in[10]; const float* bo  = (const float*)d_in[11];
    const float* g1 = (const float*)d_in[12]; const float* be1 = (const float*)d_in[13];
    const float* W1 = (const float*)d_in[14]; const float* b1  = (const float*)d_in[15];
    const float* W2 = (const float*)d_in[16]; const float* b2  = (const float*)d_in[17];
    const float* g2 = (const float*)d_in[18]; const float* be2 = (const float*)d_in[19];
    float* out = (float*)d_out;

    // bf16 workspace regions
    u16* xb    = (u16*)d_ws;
    u16* WqT   = xb    + 4194304;
    u16* WkT   = WqT   + 1048576;
    u16* WvT   = WkT   + 1048576;
    u16* WoT   = WvT   + 1048576;
    u16* W1T   = WoT   + 1048576;    // [4096][1024]
    u16* W2T   = W1T   + 4194304;    // [1024][4096]
    u16* Qb    = W2T   + 4194304;
    u16* Kbf   = Qb    + 4194304;
    u16* Vbf   = Kbf   + 4194304;
    u16* ctxb  = Vbf   + 4194304;
    u16* out1b = ctxb  + 4194304;
    u16* ff1b  = out1b + 4194304;    // [4096][4096]
    // f32 workspace regions
    float* proj  = (float*)(ff1b + 16777216);
    float* out1f = proj  + 4194304;
    float* ff2   = out1f + 4194304;

    dim3 blk(256);
    dim3 gN1024(1024 / 128, 4096 / 128);   // (8, 32)
    dim3 gN4096(4096 / 128, 4096 / 128);   // (32, 32)

    // one-time operand preparation (bf16, weights pre-transposed N-major)
    cvt_kernel<<<dim3(16384), blk, 0, stream>>>(x, xb, 4194304);
    tr_kernel<<<dim3(32, 32),  blk, 0, stream>>>(Wq, WqT, 1024, 1024);
    tr_kernel<<<dim3(32, 32),  blk, 0, stream>>>(Wk, WkT, 1024, 1024);
    tr_kernel<<<dim3(32, 32),  blk, 0, stream>>>(Wv, WvT, 1024, 1024);
    tr_kernel<<<dim3(32, 32),  blk, 0, stream>>>(Wo, WoT, 1024, 1024);
    tr_kernel<<<dim3(128, 32), blk, 0, stream>>>(W1, W1T, 1024, 4096);
    tr_kernel<<<dim3(32, 128), blk, 0, stream>>>(W2, W2T, 4096, 1024);

    // QKV projections (bf16 outputs feed attention directly)
    gemm_bf16_kernel<<<gN1024, blk, 0, stream>>>(xb, WqT, bq, nullptr, Qb,  4096, 1024, 1024, 0);
    gemm_bf16_kernel<<<gN1024, blk, 0, stream>>>(xb, WkT, bk, nullptr, Kbf, 4096, 1024, 1024, 0);
    gemm_bf16_kernel<<<gN1024, blk, 0, stream>>>(xb, WvT, bv, nullptr, Vbf, 4096, 1024, 1024, 0);

    // fused dual-map masked attention -> bf16 context
    attn_kernel<<<dim3(16, 64), blk, 0, stream>>>(Qb, Kbf, Vbf, rel, adjm, comm, ctxb);

    // output projection + residual LN1 (emits f32 for LN2 residual + bf16 for FF1)
    gemm_bf16_kernel<<<gN1024, blk, 0, stream>>>(ctxb, WoT, bo, proj, nullptr, 4096, 1024, 1024, 0);
    ln_residual_kernel<<<dim3(4096), blk, 0, stream>>>(proj, x, g1, be1, out1f, out1b);

    // FFN
    gemm_bf16_kernel<<<gN4096, blk, 0, stream>>>(out1b, W1T, b1, nullptr, ff1b, 4096, 4096, 1024, 1);
    gemm_bf16_kernel<<<gN1024, blk, 0, stream>>>(ff1b, W2T, b2, ff2, nullptr, 4096, 1024, 4096, 0);
    ln_residual_kernel<<<dim3(4096), blk, 0, stream>>>(ff2, out1f, g2, be2, out, nullptr);
}
